// MultiHeadAttention_17566416241476
// MI455X (gfx1250) — compile-verified, matrix operand order checked by reading, not run on hardware
//
#include <hip/hip_runtime.h>

typedef _Float16 F16;
typedef __attribute__((ext_vector_type(16))) _Float16 v16h;
typedef __attribute__((ext_vector_type(8)))  _Float16 v8h;
typedef __attribute__((ext_vector_type(4)))  _Float16 v4h;
typedef __attribute__((ext_vector_type(8)))  float    v8f;
typedef __attribute__((ext_vector_type(4)))  float    v4f;
typedef __attribute__((ext_vector_type(4)))  unsigned int v4u;
typedef __attribute__((ext_vector_type(8)))  int      v8i;
typedef __attribute__((ext_vector_type(4)))  int      v4i;

// ---------------------------------------------------------------------------
// NT-layout 16x32 f16 fragment loader (A: row-major MxK / B: row-major NxK).
// lane L<16: row M=L, K chunks [k..k+7], [k+16..k+23];
// lane L>=16: same row, chunks [k+8..k+15], [k+24..k+31].
// Works on global or LDS pointers (addrspace inferred after inlining).
// ---------------------------------------------------------------------------
__device__ __forceinline__ v16h load_frag(const F16* p0, int ld, int lane, int kbase) {
  const F16* p = p0 + (size_t)(lane & 15) * ld + kbase + ((lane >> 4) << 3);
  v8h lo = *(const v8h*)p;
  v8h hi = *(const v8h*)(p + 16);
  v16h r;
#pragma unroll
  for (int i = 0; i < 8; ++i) { r[i] = lo[i]; r[i + 8] = hi[i]; }
  return r;
}

__device__ __forceinline__ v8f wmma_f16(v16h a, v16h b, v8f c) {
  return __builtin_amdgcn_wmma_f32_16x16x32_f16(false, a, false, b, (short)0, c,
                                                false, false);
}

// ---------------------------------------------------------------------------
// Tensor Data Mover: async 2D tile (16-bit elements) global -> LDS.
// Builds the D# per CDNA5 ISA ch.8 and issues TENSOR_LOAD_TO_LDS.
// ---------------------------------------------------------------------------
__device__ __forceinline__ unsigned lds_off(const void* p) {
  return (unsigned)(unsigned long long)p;   // low 32 bits of LDS aperture addr
}

__device__ __forceinline__ void tdm_load_2d_f16(
    unsigned ldsAddr,          // byte offset in LDS
    const F16* gtile,          // global address of tile start
    unsigned tileD0,           // elements per row of tile
    unsigned tileD1,           // rows of tile
    unsigned tensorD0,         // tensor row length (elements)
    unsigned tensorD1,         // tensor rows
    unsigned long long stride0 // elements between consecutive rows
) {
  unsigned long long ga = (unsigned long long)gtile;
  v4u g0;
  g0[0] = 1u;                                  // count=1, user descriptor
  g0[1] = ldsAddr;                             // lds_addr
  g0[2] = (unsigned)ga;                        // global_addr[31:0]
  g0[3] = (unsigned)(ga >> 32) | (2u << 30);   // global_addr[56:32] | type=2
  v8i g1;
  g1[0] = (int)(1u << 16);                     // data_size = 1 (2 bytes/elem)
  g1[1] = (int)((tensorD0 & 0xFFFFu) << 16);   // tensor_dim0[15:0]
  g1[2] = (int)((tensorD0 >> 16) | ((tensorD1 & 0xFFFFu) << 16));
  g1[3] = (int)((tensorD1 >> 16) | (tileD0 << 16));   // tile_dim0
  g1[4] = (int)(tileD1 & 0xFFFFu);             // tile_dim1 (tile_dim2 = 0)
  g1[5] = (int)(stride0 & 0xFFFFFFFFull);      // tensor_dim0_stride[31:0]
  g1[6] = (int)((stride0 >> 32) & 0xFFFFull);  // stride[47:32] (dim1_stride=0)
  g1[7] = 0;
  v4i g2 = {0, 0, 0, 0};
  v4i g3 = {0, 0, 0, 0};
#if __has_include(<hip/amd_detail/amd_gfx1250_TDM.h>)
  v8i g4 = {0, 0, 0, 0, 0, 0, 0, 0};
  __builtin_amdgcn_tensor_load_to_lds(g0, g1, g2, g3, g4, 0);
#else
  __builtin_amdgcn_tensor_load_to_lds(g0, g1, g2, g3, 0);
#endif
}

// ---------------------------------------------------------------------------
// fp32 -> f16 conversion, 4 elements per thread
// ---------------------------------------------------------------------------
__global__ void __launch_bounds__(256) cvt_f32_f16(const v4f* __restrict__ in,
                                                   v4h* __restrict__ out, int n4) {
  int i = blockIdx.x * blockDim.x + threadIdx.x;
  if (i < n4) {
    v4f v = in[i];
    v4h h;
    h[0] = (F16)v[0]; h[1] = (F16)v[1]; h[2] = (F16)v[2]; h[3] = (F16)v[3];
    out[i] = h;
  }
}

// ---------------------------------------------------------------------------
// GEMM: C[M,N] = A[M,K] @ B[N,K]^T + bias[N]  (f16 operands, f32 accumulate)
// 256 threads = 8 waves; block tile 128x64; wave tile 16x64.
// BOTH operand k-chunks staged through double-buffered LDS by TDM:
//   A tile 128x64 f16 (16KB) + B tile 64x64 f16 (8KB), x2 buffers = 48KB.
// Inner loop is pure ds_load + v_wmma; all global traffic is async DMA.
// mode 0: f16 head-major [B,16,S,64], scaled 0.125 (Q)
// mode 1: f16 head-major [B,16,S,64]                (K)
// mode 2: f16 transposed head-major [B,16,64,S]     (V^T)
// mode 3: f32 row-major [M,N] -> outF               (final projection)
// ---------------------------------------------------------------------------
__global__ void __launch_bounds__(256) gemm_nt(
    const F16* __restrict__ A, const F16* __restrict__ Bw,
    const float* __restrict__ bias, float* __restrict__ outF,
    F16* __restrict__ outH, int M, int N, int K, int mode) {
  const int lane = threadIdx.x & 31;
  const int wave = threadIdx.x >> 5;
  const int half = lane >> 4;
  const int nl   = lane & 15;
  const int tilesN = N >> 6;
  const int bm = blockIdx.x / tilesN;
  const int bn = blockIdx.x % tilesN;
  const int row0 = bm * 128 + wave * 16;
  const int col0 = bn * 64;

  __shared__ __align__(16) F16 At[2][128][64];
  __shared__ __align__(16) F16 Bt[2][64][64];

  const F16* Atile = A  + (size_t)(bm * 128) * K;  // 128 rows x K
  const F16* Btile = Bw + (size_t)col0 * K;        // 64 rows x K

  if (wave == 0) {
    tdm_load_2d_f16(lds_off(&At[0][0][0]), Atile, 64, 128, (unsigned)K,
                    (unsigned)M, (unsigned long long)K);
    tdm_load_2d_f16(lds_off(&Bt[0][0][0]), Btile, 64, 64, (unsigned)K,
                    (unsigned)N, (unsigned long long)K);
  }

  v8f acc[4] = {};
  const int nChunks = K >> 6;
  for (int kc = 0; kc < nChunks; ++kc) {
    const int cur = kc & 1;
    if (wave == 0) {
      if (kc + 1 < nChunks) {
        tdm_load_2d_f16(lds_off(&At[cur ^ 1][0][0]), Atile + (kc + 1) * 64,
                        64, 128, (unsigned)K, (unsigned)M, (unsigned long long)K);
        tdm_load_2d_f16(lds_off(&Bt[cur ^ 1][0][0]), Btile + (kc + 1) * 64,
                        64, 64, (unsigned)K, (unsigned)N, (unsigned long long)K);
        __builtin_amdgcn_s_wait_tensorcnt(2);   // chunk kc's two tiles landed
      } else {
        __builtin_amdgcn_s_wait_tensorcnt(0);
      }
    }
    __syncthreads();

#pragma unroll
    for (int kd = 0; kd < 2; ++kd) {
      v16h a = load_frag(&At[cur][wave * 16][0], 64, lane, kd * 32);
#pragma unroll
      for (int t = 0; t < 4; ++t) {
        v16h b = load_frag(&Bt[cur][t * 16][0], 64, lane, kd * 32);
        acc[t] = wmma_f16(a, b, acc[t]);
      }
    }
    __syncthreads();
  }

  // C layout: VGPR r, lane L -> (M = r + 8*(L>=16), N = L&15)
#pragma unroll
  for (int t = 0; t < 4; ++t) {
#pragma unroll
    for (int r = 0; r < 8; ++r) {
      int m = row0 + r + half * 8;
      int n = col0 + t * 16 + nl;
      float v = acc[t][r] + bias[n];
      if (mode == 3) {
        outF[(size_t)m * N + n] = v;
      } else {
        int bb = m >> 11, s = m & 2047;   // M = 4*2048
        int hh = n >> 6,  d = n & 63;     // N = 16*64
        if (mode == 2) {
          outH[((((size_t)bb * 16 + hh) * 64 + d) * 2048) + s] = (F16)v;
        } else {
          float sv = (mode == 0) ? v * 0.125f : v;  // fold 1/sqrt(64) into Q
          outH[((((size_t)bb * 16 + hh) * 2048 + s) * 64) + d] = (F16)sv;
        }
      }
    }
  }
}

// ---------------------------------------------------------------------------
// Flash attention: one block per (b, h, 128-row query tile); 8 waves.
// K tile [64kv x 64d] and V^T tile [64d x 64kv] TDM-staged, double-buffered,
// shared by all 8 waves. Online softmax; P transposed through LDS.
// ---------------------------------------------------------------------------
__global__ void __launch_bounds__(256) flash_attn(
    const F16* __restrict__ Qh, const F16* __restrict__ Kh,
    const F16* __restrict__ Vt, F16* __restrict__ attnO) {
  constexpr int S = 2048;
  constexpr int D = 64;
  const int lane = threadIdx.x & 31;
  const int wave = threadIdx.x >> 5;
  const int half = lane >> 4;
  const int nl   = lane & 15;

  const int qt = blockIdx.x & 15;   // S/128 = 16 query tiles
  const int bh = blockIdx.x >> 4;   // 0..63 (b*16+h)
  const int b  = bh >> 4;
  const int h  = bh & 15;

  const F16* Qb = Qh + (size_t)bh * S * D;
  const F16* Kb = Kh + (size_t)bh * S * D;
  const F16* Vb = Vt + (size_t)bh * D * S;
  const int qrow0 = qt * 128 + wave * 16;

  __shared__ __align__(16) F16 Kt[2][64][64];   // [kv-rel][d]
  __shared__ __align__(16) F16 Vs[2][64][64];   // [d][kv-rel]
  __shared__ __align__(16) F16 Plds[8][16][64];

  if (wave == 0) {
    tdm_load_2d_f16(lds_off(&Kt[0][0][0]), Kb, 64, 64, D, S, (unsigned long long)D);
    tdm_load_2d_f16(lds_off(&Vs[0][0][0]), Vb, 64, 64, S, D, (unsigned long long)S);
  }

  // Q A-fragments resident for the whole kernel (pre-scaled by 1/8)
  v16h aQ[2];
  aQ[0] = load_frag(Qb + (size_t)qrow0 * D, D, lane, 0);
  aQ[1] = load_frag(Qb + (size_t)qrow0 * D, D, lane, 32);

  v8f o[4] = {};
  float mrow[8], lrow[8];
#pragma unroll
  for (int r = 0; r < 8; ++r) { mrow[r] = -1e30f; lrow[r] = 0.0f; }

  constexpr int nChunks = S / 64;
  for (int i = 0; i < nChunks; ++i) {
    const int cur = i & 1;
    const int kv  = i * 64;
    if (wave == 0) {
      if (i + 1 < nChunks) {
        tdm_load_2d_f16(lds_off(&Kt[cur ^ 1][0][0]), Kb + (size_t)(kv + 64) * D,
                        64, 64, D, S, (unsigned long long)D);
        tdm_load_2d_f16(lds_off(&Vs[cur ^ 1][0][0]), Vb + (kv + 64),
                        64, 64, S, D, (unsigned long long)S);
        __builtin_amdgcn_s_wait_tensorcnt(2);   // chunk i's two tiles landed
      } else {
        __builtin_amdgcn_s_wait_tensorcnt(0);
      }
    }
    __syncthreads();

    // ---- S = Q @ K^T (scaled) : 16x64 per wave, K from LDS ----
    v8f s[4] = {};
#pragma unroll
    for (int t = 0; t < 4; ++t) {
#pragma unroll
      for (int kd = 0; kd < 2; ++kd) {
        v16h bK = load_frag(&Kt[cur][t * 16][0], 64, lane, kd * 32);
        s[t] = wmma_f16(aQ[kd], bK, s[t]);
      }
    }

    // ---- online softmax (row stats per accumulator register r) ----
#pragma unroll
    for (int r = 0; r < 8; ++r) {
      float mx = fmaxf(fmaxf(s[0][r], s[1][r]), fmaxf(s[2][r], s[3][r]));
#pragma unroll
      for (int off = 8; off >= 1; off >>= 1)
        mx = fmaxf(mx, __shfl_xor(mx, off, 32));     // 16-lane group reduce
      float mnew = fmaxf(mrow[r], mx);
      float fs = __expf(mrow[r] - mnew);
      mrow[r] = mnew;
      float psum = 0.0f;
#pragma unroll
      for (int t = 0; t < 4; ++t) {
        float pv = __expf(s[t][r] - mnew);
        s[t][r] = pv;
        psum += pv;
      }
#pragma unroll
      for (int off = 8; off >= 1; off >>= 1)
        psum += __shfl_xor(psum, off, 32);
      lrow[r] = lrow[r] * fs + psum;
#pragma unroll
      for (int t = 0; t < 4; ++t) o[t][r] *= fs;
    }

    // ---- transpose P (C layout -> A layout) through LDS ----
#pragma unroll
    for (int t = 0; t < 4; ++t)
#pragma unroll
      for (int r = 0; r < 8; ++r)
        Plds[wave][r + half * 8][t * 16 + nl] = (F16)s[t][r];
    __syncthreads();

    // ---- O += P @ V  (V^T tile from LDS) ----
#pragma unroll
    for (int kd = 0; kd < 2; ++kd) {
      v16h aP = load_frag(&Plds[wave][0][0], 64, lane, kd * 32);
#pragma unroll
      for (int t = 0; t < 4; ++t) {
        v16h bV = load_frag(&Vs[cur][t * 16][0], 64, lane, kd * 32);
        o[t] = wmma_f16(aP, bV, o[t]);
      }
    }
  }

  // ---- epilogue: normalize, write merged-head f16 [B, S, 1024] ----
#pragma unroll
  for (int r = 0; r < 8; ++r) {
    float inv = 1.0f / lrow[r];
    int q = qrow0 + r + half * 8;
    size_t rowbase = ((size_t)b * S + q) * 1024 + h * 64;
#pragma unroll
    for (int t = 0; t < 4; ++t)
      attnO[rowbase + t * 16 + nl] = (F16)(o[t][r] * inv);
  }
}

// ---------------------------------------------------------------------------
// Host-side orchestration
// ---------------------------------------------------------------------------
extern "C" void kernel_launch(void* const* d_in, const int* in_sizes, int n_in,
                              void* d_out, int out_size, void* d_ws, size_t ws_size,
                              hipStream_t stream) {
  (void)in_sizes; (void)n_in; (void)out_size; (void)ws_size;
  const float* x  = (const float*)d_in[0];
  const float* Wq = (const float*)d_in[1];
  const float* bq = (const float*)d_in[2];
  const float* Wk = (const float*)d_in[3];
  const float* bk = (const float*)d_in[4];
  const float* Wv = (const float*)d_in[5];
  const float* bv = (const float*)d_in[6];
  const float* Wo = (const float*)d_in[7];
  const float* bo = (const float*)d_in[8];
  float* out = (float*)d_out;

  const size_t NX = (size_t)8192 * 1024;  // activation elems
  const size_t NW = (size_t)1024 * 1024;  // per-weight elems

  F16* ws  = (F16*)d_ws;
  F16* xh  = ws;              // 8M f16
  F16* Wqh = xh  + NX;        // 1M
  F16* Wkh = Wqh + NW;
  F16* Wvh = Wkh + NW;
  F16* Woh = Wvh + NW;
  F16* Qh  = Woh + NW;        // 8M (head-major, pre-scaled)
  F16* Kh  = Qh  + NX;        // 8M
  F16* Vt  = Kh  + NX;        // 8M (transposed per head)
  F16* Ah  = Vt  + NX;        // 8M (merged-head attention output)

  cvt_f32_f16<<<(int)(NX / 4 / 256), 256, 0, stream>>>((const v4f*)x, (v4h*)xh, (int)(NX / 4));
  cvt_f32_f16<<<(int)(NW / 4 / 256), 256, 0, stream>>>((const v4f*)Wq, (v4h*)Wqh, (int)(NW / 4));
  cvt_f32_f16<<<(int)(NW / 4 / 256), 256, 0, stream>>>((const v4f*)Wk, (v4h*)Wkh, (int)(NW / 4));
  cvt_f32_f16<<<(int)(NW / 4 / 256), 256, 0, stream>>>((const v4f*)Wv, (v4h*)Wvh, (int)(NW / 4));
  cvt_f32_f16<<<(int)(NW / 4 / 256), 256, 0, stream>>>((const v4f*)Wo, (v4h*)Woh, (int)(NW / 4));

  // projections: grid = (8192/128) * (1024/64) = 1024 blocks, 256 threads
  gemm_nt<<<1024, 256, 0, stream>>>(xh, Wqh, bq, nullptr, Qh, 8192, 1024, 1024, 0);
  gemm_nt<<<1024, 256, 0, stream>>>(xh, Wkh, bk, nullptr, Kh, 8192, 1024, 1024, 1);
  gemm_nt<<<1024, 256, 0, stream>>>(xh, Wvh, bv, nullptr, Vt, 8192, 1024, 1024, 2);

  // attention: B*H*(S/128) = 4*16*16 = 1024 blocks, 256 threads
  flash_attn<<<1024, 256, 0, stream>>>(Qh, Kh, Vt, Ah);

  // output projection -> f32 d_out
  gemm_nt<<<1024, 256, 0, stream>>>(Ah, Woh, bo, out, nullptr, 8192, 1024, 1024, 3);
}
